// VoxelMemory_54606214202068
// MI455X (gfx1250) — compile-verified, matrix operand order checked by reading, not run on hardware
//
#include <hip/hip_runtime.h>
#include <math.h>

// Problem constants (match reference)
#define BS     8192      // B*S
#define DIM    256       // embed dim
#define NV     16384     // voxels
#define TOPK   32
#define CHUNKS 4         // voxel-axis split for occupancy on the score phase
#define NVC    (NV / CHUNKS)

#if defined(__has_builtin)
#if __has_builtin(__builtin_amdgcn_sched_barrier)
#define SCHED_BARRIER() __builtin_amdgcn_sched_barrier(0)
#endif
#endif
#ifndef SCHED_BARRIER
#define SCHED_BARRIER()
#endif

typedef __attribute__((ext_vector_type(16))) __bf16 v16bf;
typedef __attribute__((ext_vector_type(8)))  float  v8f;

__device__ __forceinline__ unsigned short f2bf(float f) {
    union { float f; unsigned int u; } c; c.f = f;
    unsigned int x = c.u;
    unsigned int r = x + 0x7FFFu + ((x >> 16) & 1u);   // round-to-nearest-even
    return (unsigned short)(r >> 16);
}

union FragU { v16bf v; uint4 u[2]; };

// 16-bit A-layout fragment (also used for B as B^T rows):
// lane group g = lane>>4: elements 0..7  <-> K = kofs..kofs+7
//                          elements 8..15 <-> K = kofs+16..kofs+23
// with kofs = kk + 8*g.  Two contiguous 16B chunks -> two b128 loads.
__device__ __forceinline__ v16bf load_frag(const unsigned short* rowp, int kofs) {
    FragU f;
    f.u[0] = *(const uint4*)(rowp + kofs);
    f.u[1] = *(const uint4*)(rowp + kofs + 16);
    return f.v;
}

// ---------------- elementwise converts ----------------
__global__ void __launch_bounds__(256) k_f32_to_bf16(const float* __restrict__ src,
                                                     unsigned short* __restrict__ dst, int n) {
    int i = blockIdx.x * 256 + threadIdx.x;
    if (i < n) dst[i] = f2bf(src[i]);
}

__global__ void __launch_bounds__(256) k_make_ve(const float* __restrict__ base,
                                                 const float* __restrict__ ovl,
                                                 const float* __restrict__ conf,
                                                 unsigned short* __restrict__ dst) {
    int i = blockIdx.x * 256 + threadIdx.x;           // over NV*DIM
    float c = conf[i >> 8];                           // DIM == 256
    dst[i] = f2bf(base[i] + c * ovl[i]);
}

// ---------------- generic WMMA GEMM: C = A @ W^T + bias ----------------
// A: [M,DIM] bf16 row-major, W: [N,DIM] bf16 row-major (B^T rows == W rows).
// One wave computes a 32x16 tile. ALL operand fragments (8 W + 16 A) are
// loaded up front behind a sched barrier so the 48 b128 loads issue as
// clauses with staged waits, instead of load->full-wait->wmma interleaving.
// EPI: 0 = f32 store, 1 = bf16 store (val*scale), 2 = exact GELU -> bf16.
template<int EPI>
__global__ void __launch_bounds__(32) k_gemm_xwt(const unsigned short* __restrict__ A,
                                                 const unsigned short* __restrict__ W,
                                                 const float* __restrict__ bias,
                                                 void* __restrict__ Cout,
                                                 int M, int N, float scale) {
    const int lane = threadIdx.x;
    const int g = lane >> 4, m = lane & 15;
    const int m0 = blockIdx.x * 32;
    const int n0 = blockIdx.y * 16;

    const unsigned short* arow0 = A + (size_t)(m0 + m) * DIM;
    const unsigned short* arow1 = arow0 + (size_t)16 * DIM;
    const unsigned short* brow  = W + (size_t)(n0 + m) * DIM;

    v16bf bfr[8], a0[8], a1[8];
#pragma unroll
    for (int c = 0; c < 8; ++c) {
        int kofs = c * 32 + 8 * g;
        bfr[c] = load_frag(brow,  kofs);
        a0[c]  = load_frag(arow0, kofs);
        a1[c]  = load_frag(arow1, kofs);
    }
    SCHED_BARRIER();

    v8f acc0 = {0.f,0.f,0.f,0.f,0.f,0.f,0.f,0.f};
    v8f acc1 = {0.f,0.f,0.f,0.f,0.f,0.f,0.f,0.f};
#pragma unroll
    for (int c = 0; c < 8; ++c) {
        acc0 = __builtin_amdgcn_wmma_f32_16x16x32_bf16(false, a0[c], false, bfr[c], (short)0, acc0, false, false);
        acc1 = __builtin_amdgcn_wmma_f32_16x16x32_bf16(false, a1[c], false, bfr[c], (short)0, acc1, false, false);
    }

    float bn = bias[n0 + m];
    const int col = n0 + m;
#pragma unroll
    for (int r = 0; r < 8; ++r) {
        float v0 = (acc0[r] + bn) * scale;
        float v1 = (acc1[r] + bn) * scale;
        int row0 = m0 + r + 8 * g;      // C layout: N = lane&15, M = r + 8*(lane>>4)
        int row1 = row0 + 16;
        if (EPI == 0) {
            ((float*)Cout)[(size_t)row0 * N + col] = v0;
            ((float*)Cout)[(size_t)row1 * N + col] = v1;
        } else if (EPI == 1) {
            ((unsigned short*)Cout)[(size_t)row0 * N + col] = f2bf(v0);
            ((unsigned short*)Cout)[(size_t)row1 * N + col] = f2bf(v1);
        } else {
            float g0 = 0.5f * v0 * (1.f + erff(v0 * 0.70710678118654752f));
            float g1 = 0.5f * v1 * (1.f + erff(v1 * 0.70710678118654752f));
            ((unsigned short*)Cout)[(size_t)row0 * N + col] = f2bf(g0);
            ((unsigned short*)Cout)[(size_t)row1 * N + col] = f2bf(g1);
        }
    }
}

// ---------------- fused scores + partial top-k ----------------
// One wave per (16-query tile, voxel chunk). Q fragments preloaded to
// registers. The K fragments are software-pipelined across tiles: the next
// tile's 16 b128 loads are issued (behind a sched barrier) before the current
// tile's WMMA chain; their waitcnt lands at the cur<-nxt register copy at the
// bottom of the loop, i.e. AFTER the wmma chain + LDS scores + top-k
// insertion, hiding the full L2 latency inside a single wave.
__global__ void __launch_bounds__(32) k_topk_part(const unsigned short* __restrict__ Q,
                                                  const unsigned short* __restrict__ Kb,
                                                  const float* __restrict__ conf,
                                                  float* __restrict__ pval,
                                                  int* __restrict__ pidx) {
    __shared__ float sc[16 * 16];
    __shared__ float tv[16 * TOPK];
    __shared__ int   ti[16 * TOPK];

    const int lane = threadIdx.x;
    const int g = lane >> 4, m = lane & 15;
    const int qt = blockIdx.x;              // query tile id
    const int ch = blockIdx.y;              // voxel chunk id
    const int q0 = qt * 16;

    v16bf qf[8];
    const unsigned short* qrow = Q + (size_t)(q0 + m) * DIM;
#pragma unroll
    for (int c = 0; c < 8; ++c) qf[c] = load_frag(qrow, c * 32 + 8 * g);

    if (lane < 16) {
#pragma unroll
        for (int j = 0; j < TOPK; ++j) { tv[lane * TOPK + j] = -INFINITY; ti[lane * TOPK + j] = 0; }
    }
    float mn = -INFINITY; int ma = 0;
    __syncthreads();

    const int nbeg = ch * NVC, nend = nbeg + NVC;

    // prologue: load first tile's K fragments
    v16bf cur[8];
    {
        const unsigned short* krow = Kb + (size_t)(nbeg + m) * DIM;
#pragma unroll
        for (int c = 0; c < 8; ++c) cur[c] = load_frag(krow, c * 32 + 8 * g);
    }

    for (int n0 = nbeg; n0 < nend; n0 += 16) {
        // prefetch next tile (clamped; redundant load on last iteration)
        int n1 = (n0 + 16 < nend) ? (n0 + 16) : nbeg;
        const unsigned short* krow2 = Kb + (size_t)(n1 + m) * DIM;
        v16bf nxt[8];
#pragma unroll
        for (int c = 0; c < 8; ++c) nxt[c] = load_frag(krow2, c * 32 + 8 * g);
        SCHED_BARRIER();

        v8f acc = {0.f,0.f,0.f,0.f,0.f,0.f,0.f,0.f};
#pragma unroll
        for (int c = 0; c < 8; ++c)
            acc = __builtin_amdgcn_wmma_f32_16x16x32_bf16(false, qf[c], false, cur[c], (short)0, acc, false, false);

        float cterm = 2.0f * conf[n0 + m];
#pragma unroll
        for (int r = 0; r < 8; ++r) sc[(r + 8 * g) * 16 + m] = acc[r] + cterm;
        __syncthreads();

        if (lane < 16) {
            for (int c2 = 0; c2 < 16; ++c2) {
                float s = sc[lane * 16 + c2];
                if (s > mn) {
                    tv[lane * TOPK + ma] = s; ti[lane * TOPK + ma] = n0 + c2;
                    mn = tv[lane * TOPK]; ma = 0;
                    for (int j = 1; j < TOPK; ++j) {
                        float t = tv[lane * TOPK + j];
                        if (t < mn) { mn = t; ma = j; }
                    }
                }
            }
        }
        __syncthreads();

        // rotate buffers; waitcnt for the prefetch lands here
#pragma unroll
        for (int c = 0; c < 8; ++c) cur[c] = nxt[c];
    }

    if (lane < 16) {
        size_t ob = ((size_t)(qt * CHUNKS + ch) * 16 + lane) * TOPK;
#pragma unroll
        for (int j = 0; j < TOPK; ++j) {
            pval[ob + j] = tv[lane * TOPK + j];
            pidx[ob + j] = ti[lane * TOPK + j];
        }
    }
}

// ---------------- merge partial top-k lists + sort + softmax ----------------
// One thread per query; its working top-32 list lives in LDS (avoids scratch
// from dynamic indexing). 128 threads/block -> 32KB LDS.
__global__ void __launch_bounds__(128) k_topk_merge(const float* __restrict__ pval,
                                                    const int* __restrict__ pidx,
                                                    float* __restrict__ wout,
                                                    int* __restrict__ iout) {
    __shared__ float ltv[128 * TOPK];
    __shared__ int   lti[128 * TOPK];
    const int t  = threadIdx.x;
    const int qq = blockIdx.x * 128 + t;    // query id 0..BS-1
    const int qt = qq >> 4, ql = qq & 15;

    float* tv = ltv + t * TOPK;
    int*   ti = lti + t * TOPK;
#pragma unroll
    for (int j = 0; j < TOPK; ++j) { tv[j] = -INFINITY; ti[j] = 0; }
    float mn = -INFINITY; int ma = 0;

    for (int ch = 0; ch < CHUNKS; ++ch) {
        size_t ob = ((size_t)(qt * CHUNKS + ch) * 16 + ql) * TOPK;
        for (int j = 0; j < TOPK; ++j) {
            float s = pval[ob + j];
            if (s > mn) {
                tv[ma] = s; ti[ma] = pidx[ob + j];
                mn = tv[0]; ma = 0;
                for (int k = 1; k < TOPK; ++k) {
                    if (tv[k] < mn) { mn = tv[k]; ma = k; }
                }
            }
        }
    }

    // selection sort descending (matches jax.lax.top_k ordering)
    for (int i = 0; i < TOPK - 1; ++i) {
        int best = i; float bv = tv[i];
        for (int j = i + 1; j < TOPK; ++j) {
            if (tv[j] > bv) { bv = tv[j]; best = j; }
        }
        if (best != i) {
            float t0 = tv[i]; tv[i] = tv[best]; tv[best] = t0;
            int   t1 = ti[i]; ti[i] = ti[best]; ti[best] = t1;
        }
    }

    float mx = tv[0];
    float ex[TOPK]; float sum = 0.f;
#pragma unroll
    for (int j = 0; j < TOPK; ++j) { ex[j] = expf(tv[j] - mx); sum += ex[j]; }
    float inv = 1.f / sum;
    size_t ob = (size_t)qq * TOPK;
#pragma unroll
    for (int j = 0; j < TOPK; ++j) { wout[ob + j] = ex[j] * inv; iout[ob + j] = ti[j]; }
}

// ---------------- gather: retrieved = sum_k w_k * v[idx_k] ----------------
__global__ void __launch_bounds__(256) k_gather(const float* __restrict__ w,
                                                const int* __restrict__ idx,
                                                const float* __restrict__ vf,
                                                unsigned short* __restrict__ retr) {
    int row = blockIdx.x, d = threadIdx.x;
    const float* wr = w + (size_t)row * TOPK;
    const int*   ir = idx + (size_t)row * TOPK;
    float acc = 0.f;
#pragma unroll 4
    for (int k = 0; k < TOPK; ++k) acc += wr[k] * vf[(size_t)ir[k] * DIM + d];
    retr[(size_t)row * DIM + d] = f2bf(acc);
}

// ---------------- residual + LayerNorm ----------------
__global__ void __launch_bounds__(256) k_ln(const float* __restrict__ query,
                                            const float* __restrict__ r2,
                                            const float* __restrict__ gamma,
                                            const float* __restrict__ beta,
                                            float* __restrict__ out) {
    __shared__ float s1[256], s2[256];
    int row = blockIdx.x, t = threadIdx.x;
    float x = query[(size_t)row * DIM + t] + r2[(size_t)row * DIM + t];
    s1[t] = x; s2[t] = x * x;
    __syncthreads();
    for (int o = 128; o > 0; o >>= 1) {
        if (t < o) { s1[t] += s1[t + o]; s2[t] += s2[t + o]; }
        __syncthreads();
    }
    float mu  = s1[0] * (1.f / DIM);
    float var = s2[0] * (1.f / DIM) - mu * mu;
    float rs  = rsqrtf(var + 1e-5f);
    out[(size_t)row * DIM + t] = (x - mu) * rs * gamma[t] + beta[t];
}

extern "C" void kernel_launch(void* const* d_in, const int* in_sizes, int n_in,
                              void* d_out, int out_size, void* d_ws, size_t ws_size,
                              hipStream_t stream) {
    const float* query = (const float*)d_in[0];
    const float* base  = (const float*)d_in[1];
    const float* ovl   = (const float*)d_in[2];
    const float* conf  = (const float*)d_in[3];
    const float* Wq  = (const float*)d_in[4];  const float* bq  = (const float*)d_in[5];
    const float* Wk  = (const float*)d_in[6];  const float* bk  = (const float*)d_in[7];
    const float* Wv  = (const float*)d_in[8];  const float* bv  = (const float*)d_in[9];
    const float* Wo1 = (const float*)d_in[10]; const float* bo1 = (const float*)d_in[11];
    const float* Wo2 = (const float*)d_in[12]; const float* bo2 = (const float*)d_in[13];
    const float* gamma = (const float*)d_in[14];
    const float* beta  = (const float*)d_in[15];

    float* out  = (float*)d_out;
    float* wout = out + (size_t)BS * DIM;          // weights region of output

    // workspace carve-out (~68 MB)
    char* ws = (char*)d_ws; size_t off = 0;
    auto carve = [&](size_t bytes) -> char* {
        char* p = ws + off; off = (off + bytes + 255) & ~(size_t)255; return p;
    };
    unsigned short* qin_bf = (unsigned short*)carve((size_t)BS * DIM * 2);
    unsigned short* wq_bf  = (unsigned short*)carve((size_t)DIM * DIM * 2);
    unsigned short* wk_bf  = (unsigned short*)carve((size_t)DIM * DIM * 2);
    unsigned short* wv_bf  = (unsigned short*)carve((size_t)DIM * DIM * 2);
    unsigned short* wo1_bf = (unsigned short*)carve((size_t)DIM * DIM * 2);
    unsigned short* wo2_bf = (unsigned short*)carve((size_t)DIM * DIM * 2);
    unsigned short* vebf   = (unsigned short*)carve((size_t)NV * DIM * 2);
    unsigned short* qbf    = (unsigned short*)carve((size_t)BS * DIM * 2);
    unsigned short* kbf    = (unsigned short*)carve((size_t)NV * DIM * 2);
    float*          vf     = (float*)         carve((size_t)NV * DIM * 4);
    unsigned short* retrbf = (unsigned short*)carve((size_t)BS * DIM * 2);
    unsigned short* hbf    = (unsigned short*)carve((size_t)BS * DIM * 2);
    float*          r2f    = (float*)         carve((size_t)BS * DIM * 4);
    int*            topidx = (int*)           carve((size_t)BS * TOPK * 4);
    float*          pval   = (float*)         carve((size_t)BS * CHUNKS * TOPK * 4);
    int*            pidx   = (int*)           carve((size_t)BS * CHUNKS * TOPK * 4);

    // 1) precision converts
    k_f32_to_bf16<<<(BS * DIM) / 256, 256, 0, stream>>>(query, qin_bf, BS * DIM);
    k_f32_to_bf16<<<(DIM * DIM) / 256, 256, 0, stream>>>(Wq,  wq_bf,  DIM * DIM);
    k_f32_to_bf16<<<(DIM * DIM) / 256, 256, 0, stream>>>(Wk,  wk_bf,  DIM * DIM);
    k_f32_to_bf16<<<(DIM * DIM) / 256, 256, 0, stream>>>(Wv,  wv_bf,  DIM * DIM);
    k_f32_to_bf16<<<(DIM * DIM) / 256, 256, 0, stream>>>(Wo1, wo1_bf, DIM * DIM);
    k_f32_to_bf16<<<(DIM * DIM) / 256, 256, 0, stream>>>(Wo2, wo2_bf, DIM * DIM);
    k_make_ve<<<(NV * DIM) / 256, 256, 0, stream>>>(base, ovl, conf, vebf);

    // 2) projections (WMMA); fold 1/sqrt(D)=1/16 into q
    k_gemm_xwt<1><<<dim3(BS / 32, DIM / 16), 32, 0, stream>>>(qin_bf, wq_bf, bq, qbf, BS, DIM, 1.0f / 16.0f);
    k_gemm_xwt<1><<<dim3(NV / 32, DIM / 16), 32, 0, stream>>>(vebf,   wk_bf, bk, kbf, NV, DIM, 1.0f);
    k_gemm_xwt<0><<<dim3(NV / 32, DIM / 16), 32, 0, stream>>>(vebf,   wv_bf, bv, vf,  NV, DIM, 1.0f);

    // 3) fused scores + partial top-32 (4 voxel chunks -> 2048 waves), then merge
    k_topk_part<<<dim3(BS / 16, CHUNKS), 32, 0, stream>>>(qbf, kbf, conf, pval, pidx);
    k_topk_merge<<<BS / 128, 128, 0, stream>>>(pval, pidx, wout, topidx);

    // 4) gather weighted v-rows
    k_gather<<<BS, 256, 0, stream>>>(wout, topidx, vf, retrbf);

    // 5) output MLP (WMMA): GELU epilogue, then second projection
    k_gemm_xwt<2><<<dim3(BS / 32, DIM / 16), 32, 0, stream>>>(retrbf, wo1_bf, bo1, hbf, BS, DIM, 1.0f);
    k_gemm_xwt<0><<<dim3(BS / 32, DIM / 16), 32, 0, stream>>>(hbf,    wo2_bf, bo2, r2f, BS, DIM, 1.0f);

    // 6) residual + LayerNorm
    k_ln<<<BS, 256, 0, stream>>>(query, r2f, gamma, beta, out);
}
